// InBatchTripletLoss_5978594476365
// MI455X (gfx1250) — compile-verified, matrix operand order checked by reading, not run on hardware
//
#include <hip/hip_runtime.h>
#include <hip/hip_bf16.h>
#include <math.h>

// Problem constants (reference: N=8192, D=1024, fp32 inputs).
#define N_ROWS 8192
#define DIM    1024
#define EPSV   1e-6f

#define TILE   128     // output tile per block (128x128)
#define KSTEP  64      // K staged per double-buffer iteration (2 x 32-wide WMMA chunks)
#define LSTR   72      // padded LDS row stride in halfs (144 B: 16B-aligned, bank stagger)

typedef __attribute__((ext_vector_type(16))) _Float16 v16h;
typedef __attribute__((ext_vector_type(8)))  float    v8f;

typedef __attribute__((address_space(3))) _Float16 lds_f16_t;

__device__ __forceinline__ float softplusf(float x) {
    // log1p(exp(x)) computed stably
    return fmaxf(x, 0.0f) + log1pf(expf(-fabsf(x)));
}

// CDNA5 async global->LDS copy (16 B per lane), tracked by ASYNCcnt.
// Bypasses the VGPR staging round-trip (no global_load+ds_store pair).
__device__ __forceinline__ void async_copy_b128(const _Float16* g, _Float16* l) {
    unsigned lds_addr = (unsigned)(uintptr_t)(lds_f16_t*)l;   // 32-bit LDS byte address
    asm volatile("global_load_async_to_lds_b128 %0, %1, off"
                 :: "v"(lds_addr), "v"(g) : "memory");
}

__device__ __forceinline__ void wait_asynccnt0() {
    asm volatile("s_wait_asynccnt 0x0" ::: "memory");
}

// ---------------------------------------------------------------------------
// Kernel 1: per-row reductions + fp16 conversion of A and B = (pos - eps).
// One block per row (256 threads = 8 wave32).
// ---------------------------------------------------------------------------
__global__ __launch_bounds__(256)
void prep_kernel(const float* __restrict__ A, const float* __restrict__ P,
                 const float* __restrict__ Ng,
                 _Float16* __restrict__ Ah, _Float16* __restrict__ Bh,
                 float* __restrict__ a2, float* __restrict__ b2,
                 float* __restrict__ dpos, float* __restrict__ dneg,
                 float* __restrict__ accum)
{
    const int row = blockIdx.x;
    const int tid = threadIdx.x;
    const size_t base = (size_t)row * DIM;

    float s_a2 = 0.f, s_b2 = 0.f, s_dp = 0.f, s_dn = 0.f;
    for (int k = tid; k < DIM; k += 256) {
        float av = A[base + k];
        float pv = P[base + k];
        float nv = Ng[base + k];
        float bv = pv - EPSV;
        s_a2 = fmaf(av, av, s_a2);
        s_b2 = fmaf(bv, bv, s_b2);
        float dp = av - pv + EPSV; s_dp = fmaf(dp, dp, s_dp);
        float dn = av - nv + EPSV; s_dn = fmaf(dn, dn, s_dn);
        Ah[base + k] = (_Float16)av;
        Bh[base + k] = (_Float16)bv;
    }

    // wave32 reduction
    #pragma unroll
    for (int off = 16; off > 0; off >>= 1) {
        s_a2 += __shfl_xor(s_a2, off, 32);
        s_b2 += __shfl_xor(s_b2, off, 32);
        s_dp += __shfl_xor(s_dp, off, 32);
        s_dn += __shfl_xor(s_dn, off, 32);
    }
    __shared__ float red[4][8];
    const int wave = tid >> 5, lane = tid & 31;
    if (lane == 0) {
        red[0][wave] = s_a2; red[1][wave] = s_b2;
        red[2][wave] = s_dp; red[3][wave] = s_dn;
    }
    __syncthreads();
    if (tid == 0) {
        float r0 = 0.f, r1 = 0.f, r2 = 0.f, r3 = 0.f;
        #pragma unroll
        for (int w = 0; w < 8; ++w) {
            r0 += red[0][w]; r1 += red[1][w]; r2 += red[2][w]; r3 += red[3][w];
        }
        a2[row]   = r0;
        b2[row]   = r1;
        dpos[row] = sqrtf(r2);
        dneg[row] = sqrtf(r3);
        if (row == 0) accum[0] = 0.0f;   // consumed only by later kernels (stream order)
    }
}

// ---------------------------------------------------------------------------
// Kernel 2: fused GEMM (v_wmma_f32_16x16x32_f16) + distance + softplus
// epilogue + block-level sum -> atomicAdd into accum.
// 256 threads = 8 waves; each wave owns a 64x32 region (4x2 WMMA tiles).
// Double-buffered LDS fed by async global->LDS DMA: one barrier + one
// s_wait_asynccnt per 64-wide K step, 16 WMMAs in between.
// ---------------------------------------------------------------------------
__global__ __launch_bounds__(256)
void gemm_softplus_kernel(const _Float16* __restrict__ Ah,
                          const _Float16* __restrict__ Bh,
                          const float* __restrict__ a2,
                          const float* __restrict__ b2,
                          const float* __restrict__ dpos,
                          float* __restrict__ accum)
{
    __shared__ __align__(16) _Float16 sA[2][TILE][LSTR];
    __shared__ __align__(16) _Float16 sB[2][TILE][LSTR];

    const int tid   = threadIdx.x;
    const int lane  = tid & 31;
    const int wave  = tid >> 5;
    const int half  = lane >> 4;     // lane 0-15 -> 0, lane 16-31 -> 1
    const int r16   = lane & 15;
    const int waveM = wave >> 2;     // 0..1  -> row offset waveM*64
    const int waveN = wave & 3;      // 0..3  -> col offset waveN*32
    const int rowBase = blockIdx.y * TILE;
    const int colBase = blockIdx.x * TILE;

    v8f acc[4][2] = {};              // zero-initialized fp32 accumulators

    // Stage one 128x64-half tile of A and of B into LDS buffer `buf`.
    // 1024 16-byte quads per matrix, 4 per thread, all via async DMA.
    auto stage = [&](int buf, int k0) {
        #pragma unroll
        for (int i = 0; i < 4; ++i) {
            int idx  = tid + i * 256;     // 0..1023
            int row  = idx >> 3;          // 0..127
            int quad = idx & 7;           // 0..7 (8 halfs each)
            async_copy_b128(Ah + (size_t)(rowBase + row) * DIM + k0 + quad * 8,
                            &sA[buf][row][quad * 8]);
            async_copy_b128(Bh + (size_t)(colBase + row) * DIM + k0 + quad * 8,
                            &sB[buf][row][quad * 8]);
        }
    };

    stage(0, 0);
    wait_asynccnt0();
    __syncthreads();

    int cur = 0;
    for (int k0 = 0; k0 < DIM; k0 += KSTEP) {
        if (k0 + KSTEP < DIM) stage(cur ^ 1, k0 + KSTEP);   // prefetch next buffer

        // Compute on current buffer: two 32-wide K chunks, 8 WMMAs each.
        #pragma unroll
        for (int kc = 0; kc < KSTEP; kc += 32) {
            union Frag { v16h v; uint4 q[2]; } afr[4], bfr[2];
            const int kb = kc + half * 8;   // ISA 16-bit A/B fragment layout
            #pragma unroll
            for (int mt = 0; mt < 4; ++mt) {
                int r = waveM * 64 + mt * 16 + r16;
                afr[mt].q[0] = *(const uint4*)&sA[cur][r][kb];
                afr[mt].q[1] = *(const uint4*)&sA[cur][r][kb + 16];
            }
            #pragma unroll
            for (int nt = 0; nt < 2; ++nt) {
                int r = waveN * 32 + nt * 16 + r16;
                bfr[nt].q[0] = *(const uint4*)&sB[cur][r][kb];
                bfr[nt].q[1] = *(const uint4*)&sB[cur][r][kb + 16];
            }
            #pragma unroll
            for (int mt = 0; mt < 4; ++mt)
                #pragma unroll
                for (int nt = 0; nt < 2; ++nt)
                    acc[mt][nt] = __builtin_amdgcn_wmma_f32_16x16x32_f16(
                        false, afr[mt].v, false, bfr[nt].v,
                        (short)0, acc[mt][nt], false, false);
        }

        wait_asynccnt0();    // next buffer's DMA (issued by this wave) is complete
        __syncthreads();     // everyone's DMA complete; reads of `cur` are done
        cur ^= 1;
    }

    // Epilogue: C layout is m = vgpr + 8*half, n = r16 within each 16x16 tile.
    float local = 0.f;
    #pragma unroll
    for (int mt = 0; mt < 4; ++mt) {
        #pragma unroll
        for (int nt = 0; nt < 2; ++nt) {
            #pragma unroll
            for (int r = 0; r < 8; ++r) {
                int gi = rowBase + waveM * 64 + mt * 16 + r + 8 * half;
                int gj = colBase + waveN * 32 + nt * 16 + r16;
                float dot = acc[mt][nt][r];
                float d2  = fmaxf(a2[gi] + b2[gj] - 2.0f * dot, 0.0f);
                float dap = sqrtf(d2 + 1e-12f);
                float sp  = softplusf(dpos[gi] - dap);
                if (gi != gj) local += sp;
            }
        }
    }
    #pragma unroll
    for (int off = 16; off > 0; off >>= 1) local += __shfl_xor(local, off, 32);
    __shared__ float wsum[8];
    if (lane == 0) wsum[wave] = local;
    __syncthreads();
    if (tid == 0) {
        float s = 0.f;
        #pragma unroll
        for (int w = 0; w < 8; ++w) s += wsum[w];
        atomicAdd(accum, s);
    }
}

// ---------------------------------------------------------------------------
// Kernel 3: add the "original negative" term and normalize.
// ---------------------------------------------------------------------------
__global__ __launch_bounds__(256)
void finalize_kernel(const float* __restrict__ dpos, const float* __restrict__ dneg,
                     const float* __restrict__ accum, float* __restrict__ out)
{
    const int tid = threadIdx.x;
    float s = 0.f;
    for (int i = tid; i < N_ROWS; i += 256) s += softplusf(dpos[i] - dneg[i]);
    #pragma unroll
    for (int off = 16; off > 0; off >>= 1) s += __shfl_xor(s, off, 32);
    __shared__ float wsum[8];
    if ((tid & 31) == 0) wsum[tid >> 5] = s;
    __syncthreads();
    if (tid == 0) {
        double tot = (double)accum[0];
        #pragma unroll
        for (int w = 0; w < 8; ++w) tot += (double)wsum[w];
        out[0] = (float)(tot / ((double)N_ROWS * (double)N_ROWS));
    }
}

// ---------------------------------------------------------------------------
// Workspace layout (needs ~33.7 MB):
//   Ah  : N*D fp16   (16 MB)
//   Bh  : N*D fp16   (16 MB)
//   a2, b2, dpos, dneg : N fp32 each (32 KB each)
//   accum : 1 fp32 (padded)
// ---------------------------------------------------------------------------
extern "C" void kernel_launch(void* const* d_in, const int* in_sizes, int n_in,
                              void* d_out, int out_size, void* d_ws, size_t ws_size,
                              hipStream_t stream) {
    (void)in_sizes; (void)n_in; (void)out_size; (void)ws_size;
    const float* A  = (const float*)d_in[0];
    const float* P  = (const float*)d_in[1];
    const float* Ng = (const float*)d_in[2];

    char* ws = (char*)d_ws;
    size_t off = 0;
    _Float16* Ah = (_Float16*)(ws + off); off += (size_t)N_ROWS * DIM * sizeof(_Float16);
    _Float16* Bh = (_Float16*)(ws + off); off += (size_t)N_ROWS * DIM * sizeof(_Float16);
    float* a2    = (float*)(ws + off); off += (size_t)N_ROWS * sizeof(float);
    float* b2    = (float*)(ws + off); off += (size_t)N_ROWS * sizeof(float);
    float* dpos  = (float*)(ws + off); off += (size_t)N_ROWS * sizeof(float);
    float* dneg  = (float*)(ws + off); off += (size_t)N_ROWS * sizeof(float);
    float* accum = (float*)(ws + off); off += 256;

    prep_kernel<<<N_ROWS, 256, 0, stream>>>(A, P, Ng, Ah, Bh, a2, b2, dpos, dneg, accum);

    dim3 grid(N_ROWS / TILE, N_ROWS / TILE);
    gemm_softplus_kernel<<<grid, 256, 0, stream>>>(Ah, Bh, a2, b2, dpos, accum);

    finalize_kernel<<<1, 256, 0, stream>>>(dpos, dneg, accum, (float*)d_out);
}
